// TransformerBlockALL_35802847379566
// MI455X (gfx1250) — matrix operand order repaired and checked by reading, not compiled, not run on hardware
//
#include <hip/hip_runtime.h>
#include <hip/hip_bf16.h>
#include <math.h>

typedef __attribute__((ext_vector_type(16))) _Float16 v16h;
typedef __attribute__((ext_vector_type(8)))  _Float16 v8h;
typedef __attribute__((ext_vector_type(8)))  float    v8f;
typedef __attribute__((ext_vector_type(4)))  float    v4f;
typedef __attribute__((ext_vector_type(4)))  unsigned u32x4;
typedef __attribute__((ext_vector_type(8)))  unsigned u32x8;

#define B_SZ   2
#define N_PTS  65536
#define D_PT   64
#define D_MD   128
#define SCALE  (0.08838834764831845f)   // 1/sqrt(128)

// ---- workspace layout ----------------------------------------------------
// f16 weights stored PACKED in B-fragment order:
//   wf[off + ((t*KC + kc)*32 + lane)*16 + e]  -> 32B contiguous per lane.
#define WOFF_FC1   0        // K=64  N=128 KC=2 T=8  -> 8192 halves (16 KB)
#define WOFF_WV    8192     // K=128 N=128 KC=4 T=8  -> 16384 halves (32 KB)
#define WOFF_WQK   24576
#define WOFF_D2    40960
#define WOFF_G1    57344
#define WOFF_G2    73728
#define WOFF_FC2   90112    // K=128 N=64  KC=4 T=4  -> 8192
#define W_TOTAL    98304
#define STAT_BYTE  262144
#define LOG_BYTE   (1u << 20)
#define GV_ELEM    ((size_t)B_SZ * N_PTS * D_MD)

// =================== Tensor Data Mover (async DMA -> LDS) ================
// 1-row 2D descriptor: data_size=8B, tile_dim0 = bytes/8.  D# groups are
// built from uniform scalars so they materialize in SGPRs (ISA 08 §8).
__device__ __forceinline__ void tdm_load_to_lds(void* lds_dst, const void* gsrc,
                                                unsigned bytes) {
  unsigned long long ga = (unsigned long long)gsrc;
  unsigned units = bytes >> 3;                       // 8-byte elements
  u32x4 g0;
  g0[0] = 1u;                                        // count=1 (valid user D#)
  g0[1] = (unsigned)(unsigned long long)lds_dst;     // lds_addr
  g0[2] = (unsigned)(ga & 0xFFFFFFFFu);              // global_addr[31:0]
  g0[3] = (unsigned)((ga >> 32) & 0x01FFFFFFu) | (2u << 30);  // addr hi | type=2
  u32x8 g1;
  g1[0] = 3u << 16;                                  // data_size = 8B
  g1[1] = (units & 0xFFFFu) << 16;                   // tensor_dim0[15:0]
  g1[2] = ((units >> 16) & 0xFFFFu) | (1u << 16);    // tensor_dim0[31:16] | dim1=1
  g1[3] = (units & 0xFFFFu) << 16;                   // tile_dim0
  g1[4] = 1u;                                        // tile_dim1 = 1
  g1[5] = units;                                     // tensor_dim0_stride
  g1[6] = 0u;
  g1[7] = 0u;
  asm volatile("tensor_load_to_lds %0, %1" :: "s"(g0), "s"(g1) : "memory");
}
__device__ __forceinline__ void tdm_wait() { __builtin_amdgcn_s_wait_tensorcnt(0); }

// ======================= WMMA fragment helpers ===========================
__device__ __forceinline__ v16h frag_a_lds(const _Float16* buf, int kbase, int lane) {
  int row = lane & 15;
  int k0  = kbase + ((lane >> 4) ? 8 : 0);
  const _Float16* p = buf + row * 128 + k0;
  v8h lo = *(const v8h*)p;
  v8h hi = *(const v8h*)(p + 16);
  return __builtin_shufflevector(lo, hi, 0, 1, 2, 3, 4, 5, 6, 7,
                                 8, 9, 10, 11, 12, 13, 14, 15);
}

__device__ __forceinline__ v16h frag_a_gf32(const float* buf, int kbase, int lane) {
  int row = lane & 15;
  int k0  = kbase + ((lane >> 4) ? 8 : 0);
  const float* p = buf + row * D_PT + k0;
  v4f x0 = *(const v4f*)p;
  v4f x1 = *(const v4f*)(p + 4);
  v4f x2 = *(const v4f*)(p + 16);
  v4f x3 = *(const v4f*)(p + 20);
  v16h a;
#pragma unroll
  for (int i = 0; i < 4; ++i) {
    a[i]      = (_Float16)x0[i];
    a[4 + i]  = (_Float16)x1[i];
    a[8 + i]  = (_Float16)x2[i];
    a[12 + i] = (_Float16)x3[i];
  }
  return a;
}

// B fragment: 32B contiguous per lane (global packed weights, L2-hot)
__device__ __forceinline__ v16h frag_b_packed(const _Float16* wf, int KC, int kc,
                                              int t, int lane) {
  return *(const v16h*)(wf + (((t * KC + kc) * 32 + lane) << 4));
}
// B fragment from LDS (TDM-staged weights): 32B contiguous -> ds_load_b128 x2
__device__ __forceinline__ v16h frag_b_lds(const _Float16* wf, int KC, int kc,
                                           int t, int lane) {
  const _Float16* p = wf + (((t * KC + kc) * 32 + lane) << 4);
  v8h lo = *(const v8h*)p;
  v8h hi = *(const v8h*)(p + 8);
  return __builtin_shufflevector(lo, hi, 0, 1, 2, 3, 4, 5, 6, 7,
                                 8, 9, 10, 11, 12, 13, 14, 15);
}

__device__ __forceinline__ v8f frag_c_bias(const float* bias, int nbase, int lane) {
  float bv = bias[nbase + (lane & 15)];
  v8f c;
#pragma unroll
  for (int j = 0; j < 8; ++j) c[j] = bv;
  return c;
}

__device__ __forceinline__ void store_c_lds_f16(_Float16* buf, int nbase,
                                                v8f c, int lane) {
  int col = nbase + (lane & 15);
  int mh  = (lane >> 4) * 8;
#pragma unroll
  for (int j = 0; j < 8; ++j) buf[(mh + j) * 128 + col] = (_Float16)c[j];
}

#define WMMA(acc, a, b) \
  (acc) = __builtin_amdgcn_wmma_f32_16x16x32_f16(false, (a), false, (b), (short)0, (acc), false, false)

__device__ __forceinline__ unsigned fkey(float f) {
  unsigned b = __float_as_uint(f);
  return (b & 0x80000000u) ? ~b : (b | 0x80000000u);
}
__device__ __forceinline__ float fdec(unsigned k) {
  unsigned b = (k & 0x80000000u) ? (k & 0x7FFFFFFFu) : ~k;
  return __uint_as_float(b);
}

// ========================= kernel 0: weight prep =========================
__device__ __forceinline__ int src_index(int tid, int K, int N) {
  int e    = tid & 15;
  int lane = (tid >> 4) & 31;
  int frag = tid >> 9;
  int KC   = K >> 5;
  int kc   = frag % KC;
  int t    = frag / KC;
  int col  = t * 16 + (lane & 15);
  int k    = kc * 32 + ((lane >> 4) ? 8 : 0) + (e < 8 ? e : e + 8);
  return k * N + col;
}

__global__ void prep_weights(const float* fc1, const float* fc2, const float* d2,
                             const float* g1, const float* g2,
                             const float* wq, const float* wk, const float* wv,
                             _Float16* wh) {
  int tid = blockIdx.x * blockDim.x + threadIdx.x;
  if (tid >= W_TOTAL) return;
  if (tid < WOFF_WV)  { wh[tid] = (_Float16)fc1[src_index(tid, 64, 128)]; return; }
  if (tid < WOFF_WQK) { int i = src_index(tid - WOFF_WV, 128, 128);
                        wh[tid] = (_Float16)wv[i]; return; }
  if (tid < WOFF_D2)  { int i = src_index(tid - WOFF_WQK, 128, 128);
                        wh[tid] = (_Float16)(wq[i] - wk[i]); return; }
  if (tid < WOFF_G1)  { int i = src_index(tid - WOFF_D2, 128, 128);
                        wh[tid] = (_Float16)d2[i]; return; }
  if (tid < WOFF_G2)  { int i = src_index(tid - WOFF_G1, 128, 128);
                        wh[tid] = (_Float16)g1[i]; return; }
  if (tid < WOFF_FC2) { int i = src_index(tid - WOFF_G2, 128, 128);
                        wh[tid] = (_Float16)g2[i]; return; }
  { int i = src_index(tid - WOFF_FC2, 128, 64); wh[tid] = (_Float16)fc2[i]; }
}

// ================== kernel 1: fused WMMA forward pass ====================
// 64 threads = 2 waves/block; per-wave 16-point tile.
// LDS: 2 waves * 3 staging bufs (24 KB) + 32 KB TDM weight buffer = 56 KB.
__global__ void __launch_bounds__(64)
fused_fwd(const float* __restrict__ xyz, const float* __restrict__ feat,
          const float* __restrict__ fc1_b, const float* __restrict__ d1_w,
          const float* __restrict__ d1_b,  const float* __restrict__ d2_b,
          const float* __restrict__ g1_b,  const float* __restrict__ g2_b,
          const _Float16* __restrict__ wh,
          float* __restrict__ logits, float* __restrict__ gatev) {
  __shared__ _Float16 smem[2 * 3 * 16 * 128];   // 24 KB staging
  __shared__ _Float16 wlds[16384];              // 32 KB weight buffer (TDM dest)
  const int lane = threadIdx.x & 31;
  const int wave = threadIdx.x >> 5;
  const int tile = blockIdx.x * 2 + wave;       // 8192 tiles (exact)
  const int b    = tile >> 12;
  const int n0   = (tile & 4095) * 16;
  const size_t gm0 = (size_t)b * N_PTS + n0;

  _Float16* xbuf = smem + wave * 3 * 2048;
  _Float16* buf1 = xbuf + 2048;
  _Float16* buf2 = buf1 + 2048;

  // Kick DMA of the first weight block; it overlaps the pos-MLP VALU work.
  if (wave == 0) tdm_load_to_lds(wlds, wh + WOFF_D2, 32768);

  // ---- pos_h = relu(xyz @ d1 + d1_b) -> buf1 ----------------------------
  const float* xz = xyz + gm0 * 3;
  for (int idx = lane; idx < 2048; idx += 32) {
    int m = idx >> 7, c = idx & 127;
    float s = d1_b[c] + xz[m * 3 + 0] * d1_w[c]
                      + xz[m * 3 + 1] * d1_w[128 + c]
                      + xz[m * 3 + 2] * d1_w[256 + c];
    buf1[idx] = (_Float16)fmaxf(s, 0.0f);
  }
  if (wave == 0) tdm_wait();
  __syncthreads();                               // pos_h + d2 weights ready

  const int col0 = lane & 15;
  const int mh   = (lane >> 4) * 8;
  v16h aF[4];

  // ---- pos = pos_h @ d2 + d2_b (resident C fragments) -------------------
  v8f posC[8];
#pragma unroll
  for (int kc = 0; kc < 4; ++kc) aF[kc] = frag_a_lds(buf1, kc * 32, lane);
#pragma unroll
  for (int t = 0; t < 8; ++t) {
    v8f acc = frag_c_bias(d2_b, t * 16, lane);
#pragma unroll
    for (int kc = 0; kc < 4; ++kc)
      WMMA(acc, aF[kc], frag_b_lds(wlds, 4, kc, t, lane));
    posC[t] = acc;
  }
  __syncthreads();                               // done reading wlds
  if (wave == 0) { tdm_load_to_lds(wlds, wh + WOFF_FC1, 16384); tdm_wait(); }
  __syncthreads();

  // ---- x = feat @ fc1 + fc1_b -> xbuf -----------------------------------
  const float* fbase = feat + gm0 * D_PT;
#pragma unroll
  for (int kc = 0; kc < 2; ++kc) aF[kc] = frag_a_gf32(fbase, kc * 32, lane);
#pragma unroll
  for (int t = 0; t < 8; ++t) {
    v8f acc = frag_c_bias(fc1_b, t * 16, lane);
#pragma unroll
    for (int kc = 0; kc < 2; ++kc)
      WMMA(acc, aF[kc], frag_b_lds(wlds, 2, kc, t, lane));
    store_c_lds_f16(xbuf, t * 16, acc, lane);
  }
  __syncthreads();
  if (wave == 0) { tdm_load_to_lds(wlds, wh + WOFF_WQK, 32768); tdm_wait(); }
  __syncthreads();

  // ---- u = x @ (wq-wk) + pos -> buf2 ------------------------------------
#pragma unroll
  for (int kc = 0; kc < 4; ++kc) aF[kc] = frag_a_lds(xbuf, kc * 32, lane);
#pragma unroll
  for (int t = 0; t < 8; ++t) {
    v8f acc = posC[t];
#pragma unroll
    for (int kc = 0; kc < 4; ++kc)
      WMMA(acc, aF[kc], frag_b_lds(wlds, 4, kc, t, lane));
    store_c_lds_f16(buf2, t * 16, acc, lane);
  }
  __syncthreads();
  if (wave == 0) { tdm_load_to_lds(wlds, wh + WOFF_G1, 32768); tdm_wait(); }
  __syncthreads();

  // ---- h = relu(u @ g1 + g1_b) -> buf1 ----------------------------------
#pragma unroll
  for (int kc = 0; kc < 4; ++kc) aF[kc] = frag_a_lds(buf2, kc * 32, lane);
#pragma unroll
  for (int t = 0; t < 8; ++t) {
    v8f acc = frag_c_bias(g1_b, t * 16, lane);
#pragma unroll
    for (int kc = 0; kc < 4; ++kc)
      WMMA(acc, aF[kc], frag_b_lds(wlds, 4, kc, t, lane));
#pragma unroll
    for (int j = 0; j < 8; ++j) acc[j] = fmaxf(acc[j], 0.0f);
    store_c_lds_f16(buf1, t * 16, acc, lane);
  }
  __syncthreads();
  if (wave == 0) { tdm_load_to_lds(wlds, wh + WOFF_G2, 32768); tdm_wait(); }
  __syncthreads();

  // ---- logits = h @ g2 + g2_b -> global ---------------------------------
#pragma unroll
  for (int kc = 0; kc < 4; ++kc) aF[kc] = frag_a_lds(buf1, kc * 32, lane);
#pragma unroll
  for (int t = 0; t < 8; ++t) {
    v8f acc = frag_c_bias(g2_b, t * 16, lane);
#pragma unroll
    for (int kc = 0; kc < 4; ++kc)
      WMMA(acc, aF[kc], frag_b_lds(wlds, 4, kc, t, lane));
    int col = t * 16 + col0;
#pragma unroll
    for (int j = 0; j < 8; ++j)
      logits[(gm0 + mh + j) * D_MD + col] = acc[j];
  }
  __syncthreads();
  if (wave == 0) { tdm_load_to_lds(wlds, wh + WOFF_WV, 32768); tdm_wait(); }
  __syncthreads();

  // ---- gatev = x @ wv + pos -> global -----------------------------------
#pragma unroll
  for (int kc = 0; kc < 4; ++kc) aF[kc] = frag_a_lds(xbuf, kc * 32, lane);
#pragma unroll
  for (int t = 0; t < 8; ++t) {
    v8f acc = posC[t];
#pragma unroll
    for (int kc = 0; kc < 4; ++kc)
      WMMA(acc, aF[kc], frag_b_lds(wlds, 4, kc, t, lane));
    int col = t * 16 + col0;
#pragma unroll
    for (int j = 0; j < 8; ++j)
      gatev[(gm0 + mh + j) * D_MD + col] = acc[j];
  }
}

// ====================== kernels 2-4: column softmax ======================
__global__ void stat_init(unsigned* maxk, float* sums) {
  int t = threadIdx.x;
  if (t < 256) { maxk[t] = fkey(-INFINITY); sums[t] = 0.0f; }
}

__global__ void __launch_bounds__(512)
col_max(const float* __restrict__ logits, unsigned* __restrict__ maxk) {
  __shared__ float red[4][128];
  const int c = threadIdx.x, ty = threadIdx.y, b = blockIdx.y;
  const int rbase = blockIdx.x * 512;
  float m = -INFINITY;
  for (int r = ty; r < 512; r += 4) {
    size_t n = (size_t)b * N_PTS + rbase + r;
    m = fmaxf(m, logits[n * D_MD + c] * SCALE);
  }
  red[ty][c] = m;
  __syncthreads();
  if (ty == 0) {
    m = fmaxf(fmaxf(red[0][c], red[1][c]), fmaxf(red[2][c], red[3][c]));
    atomicMax(&maxk[b * D_MD + c], fkey(m));
  }
}

__global__ void __launch_bounds__(512)
col_sum(const float* __restrict__ logits, const unsigned* __restrict__ maxk,
        float* __restrict__ sums) {
  __shared__ float red[4][128];
  const int c = threadIdx.x, ty = threadIdx.y, b = blockIdx.y;
  const int rbase = blockIdx.x * 512;
  const float mx = fdec(maxk[b * D_MD + c]);
  float s = 0.0f;
  for (int r = ty; r < 512; r += 4) {
    size_t n = (size_t)b * N_PTS + rbase + r;
    s += __expf(logits[n * D_MD + c] * SCALE - mx);
  }
  red[ty][c] = s;
  __syncthreads();
  if (ty == 0) {
    s = red[0][c] + red[1][c] + red[2][c] + red[3][c];
    atomicAdd(&sums[b * D_MD + c], s);
  }
}

// ================= kernel 5: softmax apply + fc2 + residual ==============
__global__ void __launch_bounds__(128)
finalize(const float* __restrict__ logits, const float* __restrict__ gatev,
         const float* __restrict__ feat, const float* __restrict__ fc2_b,
         const _Float16* __restrict__ wh,
         const unsigned* __restrict__ maxk, const float* __restrict__ sums,
         float* __restrict__ out_res, float* __restrict__ out_attn) {
  __shared__ _Float16 gbuf_all[4 * 16 * 128];   // 16 KB
  const int lane = threadIdx.x & 31;
  const int wave = threadIdx.x >> 5;
  const int tile = blockIdx.x * 4 + wave;
  const int b    = tile >> 12;
  const int n0   = (tile & 4095) * 16;
  const size_t gm0 = (size_t)b * N_PTS + n0;
  _Float16* gbuf = gbuf_all + wave * 2048;

  __builtin_prefetch(wh + WOFF_FC2, 0, 1);

  // attn = exp(z-max)/sum ; g = attn*gatev  -- coalesced row-major mapping
#pragma unroll
  for (int i = 0; i < 64; ++i) {
    int idx = i * 32 + lane;                 // [16][128] row-major
    int c   = idx & 127;
    size_t g = gm0 * D_MD + idx;
    float mx  = fdec(maxk[b * D_MD + c]);
    float inv = 1.0f / sums[b * D_MD + c];
    float a = __expf(logits[g] * SCALE - mx) * inv;
    out_attn[g] = a;
    gbuf[idx] = (_Float16)(a * gatev[g]);
  }
  __syncthreads();

  // res = g @ fc2 + fc2_b + features
  const int col0 = lane & 15;
  const int mh   = (lane >> 4) * 8;
  const float* fbase = feat + gm0 * D_PT;
  v16h aF[4];
#pragma unroll
  for (int kc = 0; kc < 4; ++kc) aF[kc] = frag_a_lds(gbuf, kc * 32, lane);
#pragma unroll
  for (int t = 0; t < 4; ++t) {
    v8f acc = frag_c_bias(fc2_b, t * 16, lane);
#pragma unroll
    for (int kc = 0; kc < 4; ++kc)
      WMMA(acc, aF[kc], frag_b_packed(wh + WOFF_FC2, 4, kc, t, lane));
    int col = t * 16 + col0;
#pragma unroll
    for (int j = 0; j < 8; ++j)
      out_res[(gm0 + mh + j) * D_PT + col] = acc[j] + fbase[(mh + j) * D_PT + col];
  }
}

// ============================== launcher =================================
extern "C" void kernel_launch(void* const* d_in, const int* in_sizes, int n_in,
                              void* d_out, int out_size, void* d_ws, size_t ws_size,
                              hipStream_t stream) {
  const float* xyz   = (const float*)d_in[0];
  const float* feat  = (const float*)d_in[1];
  const float* fc1_w = (const float*)d_in[2];
  const float* fc1_b = (const float*)d_in[3];
  const float* fc2_w = (const float*)d_in[4];
  const float* fc2_b = (const float*)d_in[5];
  const float* d1_w  = (const float*)d_in[6];
  const float* d1_b  = (const float*)d_in[7];
  const float* d2_w  = (const float*)d_in[8];
  const float* d2_b  = (const float*)d_in[9];
  const float* g1_w  = (const float*)d_in[10];
  const float* g1_b  = (const float*)d_in[11];
  const float* g2_w  = (const float*)d_in[12];
  const float* g2_b  = (const float*)d_in[13];
  const float* wq    = (const float*)d_in[14];
  const float* wk    = (const float*)d_in[15];
  const float* wv    = (const float*)d_in[16];

  char* ws = (char*)d_ws;
  _Float16* wh   = (_Float16*)ws;
  unsigned* maxk = (unsigned*)(ws + STAT_BYTE);
  float*    sums = (float*)(ws + STAT_BYTE + 1024);
  float*  logits = (float*)(ws + LOG_BYTE);
  float*  gatev  = logits + GV_ELEM;

  float* out_res  = (float*)d_out;                          // [B,N,64]
  float* out_attn = out_res + (size_t)B_SZ * N_PTS * D_PT;  // [B,N,128]

  prep_weights<<<(W_TOTAL + 255) / 256, 256, 0, stream>>>(
      fc1_w, fc2_w, d2_w, g1_w, g2_w, wq, wk, wv, wh);

  fused_fwd<<<4096, 64, 0, stream>>>(xyz, feat, fc1_b, d1_w, d1_b, d2_b,
                                     g1_b, g2_b, wh, logits, gatev);

  stat_init<<<1, 256, 0, stream>>>(maxk, sums);
  col_max<<<dim3(N_PTS / 512, B_SZ), dim3(128, 4), 0, stream>>>(logits, maxk);
  col_sum<<<dim3(N_PTS / 512, B_SZ), dim3(128, 4), 0, stream>>>(logits, maxk, sums);

  finalize<<<2048, 128, 0, stream>>>(logits, gatev, feat, fc2_b, wh,
                                     maxk, sums, out_res, out_attn);
}
